// PointNetAttnEncoder_49237505082107
// MI455X (gfx1250) — matrix operand order repaired and checked by reading, not compiled, not run on hardware
//
#include <hip/hip_runtime.h>
#include <hip/hip_bf16.h>

typedef __attribute__((ext_vector_type(16))) _Float16 v16h;
typedef __attribute__((ext_vector_type(8)))  _Float16 h8;
typedef __attribute__((ext_vector_type(8)))  float    v8f;

// ---------------- problem constants ----------------
#define BB   16
#define NN   8192
#define S1   512
#define NS1  32
#define S2   128
#define NS2  64

// ============================================================
// split xyz (B,6,N) -> coords (B,N,3) and sim (B,N,3)
// ============================================================
__global__ void k_split(const float* __restrict__ xyz,
                        float* __restrict__ coords, float* __restrict__ sim) {
  int gid = blockIdx.x * blockDim.x + threadIdx.x;
  if (gid >= BB * NN) return;
  int b = gid / NN, n = gid % NN;
  const float* src = xyz + (size_t)b * 6 * NN + n;
  float* c = coords + (size_t)gid * 3;
  float* s = sim + (size_t)gid * 3;
  c[0] = src[0];        c[1] = src[(size_t)NN];     c[2] = src[2 * (size_t)NN];
  s[0] = src[3 * (size_t)NN]; s[1] = src[4 * (size_t)NN]; s[2] = src[5 * (size_t)NN];
}

// ============================================================
// farthest point sampling: one block per batch, sequential K
// ============================================================
#define FPST 256
__global__ void k_fps(const float* __restrict__ xyz, // B x N x 3
                      float* __restrict__ dists,     // B x N scratch
                      int* __restrict__ out_idx,     // B x K
                      int N, int K) {
  int b = blockIdx.x;
  const float* px = xyz + (size_t)b * N * 3;
  float* dd = dists + (size_t)b * N;
  __shared__ float s_val[FPST];
  __shared__ int   s_idx[FPST];
  __shared__ int   s_far;
  __shared__ float s_c[3];
  int t = threadIdx.x;
  for (int i = t; i < N; i += FPST) dd[i] = 1e10f;
  if (t == 0) s_far = 0;
  __syncthreads();
  for (int it = 0; it < K; ++it) {
    if (t == 0) {
      int far = s_far;
      out_idx[(size_t)b * K + it] = far;
      s_c[0] = px[far * 3 + 0]; s_c[1] = px[far * 3 + 1]; s_c[2] = px[far * 3 + 2];
    }
    __syncthreads();
    float cx = s_c[0], cy = s_c[1], cz = s_c[2];
    float bestv = -1.0f; int besti = N - 1;
    for (int i = t; i < N; i += FPST) {
      float dx = px[i * 3] - cx, dy = px[i * 3 + 1] - cy, dz = px[i * 3 + 2] - cz;
      float d = dx * dx + dy * dy + dz * dz;
      float m = dd[i]; m = d < m ? d : m; dd[i] = m;
      if (m > bestv) { bestv = m; besti = i; }
    }
    s_val[t] = bestv; s_idx[t] = besti;
    __syncthreads();
    for (int s = FPST / 2; s > 0; s >>= 1) {
      if (t < s) {
        if (s_val[t + s] > s_val[t] ||
            (s_val[t + s] == s_val[t] && s_idx[t + s] < s_idx[t])) {
          s_val[t] = s_val[t + s]; s_idx[t] = s_idx[t + s];
        }
      }
      __syncthreads();
    }
    if (t == 0) s_far = s_idx[0];
    __syncthreads();
  }
}

// gather 3-vectors by index: dst[b,s,:] = src[b, idx[b,s], :]
__global__ void k_gather3(const float* __restrict__ src, const int* __restrict__ idx,
                          float* __restrict__ dst, int N, int S) {
  int gid = blockIdx.x * blockDim.x + threadIdx.x;
  if (gid >= BB * S) return;
  int b = gid / S;
  int pi = idx[gid];
  const float* p = src + ((size_t)b * N + pi) * 3;
  float* o = dst + (size_t)gid * 3;
  o[0] = p[0]; o[1] = p[1]; o[2] = p[2];
}

// ============================================================
// ball query: first K points (by index) within radius; pad w/ first
// ============================================================
__global__ void k_ball(const float* __restrict__ new_xyz, const float* __restrict__ xyz,
                       int* __restrict__ out, int S, int N, int K, float r2) {
  int gid = blockIdx.x * blockDim.x + threadIdx.x;
  if (gid >= BB * S) return;
  int b = gid / S;
  const float* q = new_xyz + (size_t)gid * 3;
  const float* p = xyz + (size_t)b * N * 3;
  int* o = out + (size_t)gid * K;
  float qx = q[0], qy = q[1], qz = q[2];
  int cnt = 0, first = 0;
  for (int i = 0; i < N && cnt < K; ++i) {
    float dx = p[i * 3] - qx, dy = p[i * 3 + 1] - qy, dz = p[i * 3 + 2] - qz;
    if (dx * dx + dy * dy + dz * dz <= r2) {
      if (cnt == 0) first = i;
      o[cnt++] = i;
    }
  }
  for (int j = cnt; j < K; ++j) o[j] = first;
}

// ============================================================
// grouped-feature builders (write f16, zero-padded rows)
// ============================================================
__global__ void k_group1(const float* __restrict__ xyz, const float* __restrict__ new_xyz,
                         const int* __restrict__ idx, _Float16* __restrict__ X,
                         int S, int N, int K, int Kpad) {
  int row = blockIdx.x * blockDim.x + threadIdx.x;
  if (row >= BB * S * K) return;
  int bs = row / K, b = bs / S;
  int pi = idx[row];
  const float* p = xyz + ((size_t)b * N + pi) * 3;
  const float* q = new_xyz + (size_t)bs * 3;
  _Float16* x = X + (size_t)row * Kpad;
  x[0] = (_Float16)(p[0] - q[0]);
  x[1] = (_Float16)(p[1] - q[1]);
  x[2] = (_Float16)(p[2] - q[2]);
  for (int c = 3; c < Kpad; ++c) x[c] = (_Float16)0.f;
}

__global__ void k_group2(const float* __restrict__ sub_xyz, const float* __restrict__ sub_pts,
                         const float* __restrict__ new_xyz, const int* __restrict__ idx,
                         _Float16* __restrict__ X, int Ssub, int S, int K, int C, int Kpad) {
  int row = blockIdx.x * blockDim.x + threadIdx.x;
  if (row >= BB * S * K) return;
  int bs = row / K, b = bs / S;
  int pi = idx[row];
  const float* p = sub_xyz + ((size_t)b * Ssub + pi) * 3;
  const float* q = new_xyz + (size_t)bs * 3;
  const float* f = sub_pts + ((size_t)b * Ssub + pi) * C;
  _Float16* x = X + (size_t)row * Kpad;
  x[0] = (_Float16)(p[0] - q[0]);
  x[1] = (_Float16)(p[1] - q[1]);
  x[2] = (_Float16)(p[2] - q[2]);
  for (int c = 0; c < C; ++c) x[3 + c] = (_Float16)f[c];
  for (int c = 3 + C; c < Kpad; ++c) x[c] = (_Float16)0.f;
}

__global__ void k_build3(const float* __restrict__ xyz, const float* __restrict__ pts,
                         _Float16* __restrict__ X, int G, int C, int Kpad) {
  int g = blockIdx.x * blockDim.x + threadIdx.x;
  if (g >= G) return;
  _Float16* x = X + (size_t)g * Kpad;
  x[0] = (_Float16)xyz[(size_t)g * 3 + 0];
  x[1] = (_Float16)xyz[(size_t)g * 3 + 1];
  x[2] = (_Float16)xyz[(size_t)g * 3 + 2];
  const float* f = pts + (size_t)g * C;
  for (int c = 0; c < C; ++c) x[3 + c] = (_Float16)f[c];
  for (int c = 3 + C; c < Kpad; ++c) x[c] = (_Float16)0.f;
}

// ============================================================
// weight pad+convert: W f32 (Nout x Kin) -> f16 (Nout x Kpad), zero pad
// ============================================================
__global__ void k_wpad(const float* __restrict__ W, _Float16* __restrict__ Wh,
                       int Nout, int Kin, int Kpad) {
  int gid = blockIdx.x * blockDim.x + threadIdx.x;
  if (gid >= Nout * Kpad) return;
  int n = gid / Kpad, k = gid % Kpad;
  Wh[gid] = (k < Kin) ? (_Float16)W[(size_t)n * Kin + k] : (_Float16)0.f;
}

// ============================================================
// WMMA GEMM fragment helpers (NT = 4 fixed)
// ============================================================
__device__ __forceinline__ void ld_frag(const _Float16* Abase,
                                        const _Float16* const* Bbase, int k0,
                                        h8& a0, h8& a1, h8* b0, h8* b1) {
  // speculative prefetch of the A stream ahead (global_prefetch_b8)
  __builtin_prefetch(Abase + k0 + 128, 0, 3);
  a0 = *(const h8*)(Abase + k0);
  a1 = *(const h8*)(Abase + k0 + 16);
#pragma unroll
  for (int j = 0; j < 4; ++j) {
    b0[j] = *(const h8*)(Bbase[j] + k0);
    b1[j] = *(const h8*)(Bbase[j] + k0 + 16);
  }
}

__device__ __forceinline__ void mma_frag(const h8& a0, const h8& a1,
                                         const h8* b0, const h8* b1, v8f* acc) {
  v16h a;
#pragma unroll
  for (int i = 0; i < 8; ++i) { a[i] = a0[i]; a[8 + i] = a1[i]; }
#pragma unroll
  for (int j = 0; j < 4; ++j) {
    v16h b;
#pragma unroll
    for (int i = 0; i < 8; ++i) { b[i] = b0[j][i]; b[8 + i] = b1[j][i]; }
    acc[j] = __builtin_amdgcn_wmma_f32_16x16x32_f16(
        /*neg_a=*/false, a, /*neg_b=*/false, b,
        /*c_mod=*/(short)0, acc[j], /*reuse_a=*/false, /*reuse_b=*/false);
  }
}

// ============================================================
// WMMA GEMM: C(MxN) = act(A(MxK,f16) * W^T(N x K,f16) + bias)
// One wave computes a 16x64 output strip (4 accumulators): A fragment
// reused 4x per k-step (A is the large streaming operand; weights stay
// cache-resident). Software-pipelined double buffering: the loads for
// k-step s+1 target a *different* fragment set than the WMMAs of step s
// consume -> no WMMA->VALU WAR hazard NOPs, and loads overlap matrix
// ops with partial s_wait_loadcnt. __launch_bounds__(256,1) gives the
// allocator headroom for both fragment sets (~140 VGPRs).
// Strip selection is wave-uniform so EXEC is all-ones at the WMMA.
// A fragment (ISA 7.12.2, 16-bit A 16x32): lane half h reads row
// m=lane&15, K octets [8h..8h+7] and [16+8h..16+8h+7] -> 2 b128 loads.
// ============================================================
__global__ __launch_bounds__(256, 1)
void k_gemm_wmma(const _Float16* __restrict__ A,
                 const _Float16* __restrict__ Wt,
                 const float* __restrict__ bias,
                 _Float16* __restrict__ Ch, float* __restrict__ Cf,
                 int M, int N, int K, int relu, int outH) {
  const int lane = threadIdx.x & 31;
  const int wave = threadIdx.x >> 5;
  const int ngrp = N >> 6;                      // strips of 4 N-tiles
  const int total = (M >> 4) * ngrp;
  int wt = blockIdx.x * (blockDim.x >> 5) + wave;
  if (wt >= total) return;                      // wave-uniform exit
  const int mt = wt / ngrp, ng = wt % ngrp;

  const int half = lane >> 4;
  const int l16  = lane & 15;
  const int mrow = (mt << 4) + l16;             // A row this lane loads

  v8f acc[4] = {v8f{}, v8f{}, v8f{}, v8f{}};
  const _Float16* Abase = A + (size_t)mrow * K + 8 * half;
  const _Float16* Bbase[4];
#pragma unroll
  for (int j = 0; j < 4; ++j)
    Bbase[j] = Wt + (size_t)((ng << 6) + (j << 4) + l16) * K + 8 * half;

  // ---- software pipeline: prime, then load(next) / mma(cur) ----
  h8 a0c, a1c, b0c[4], b1c[4];
  ld_frag(Abase, Bbase, 0, a0c, a1c, b0c, b1c);
#pragma unroll 2
  for (int k0 = 32; k0 < K; k0 += 32) {
    h8 a0n, a1n, b0n[4], b1n[4];
    ld_frag(Abase, Bbase, k0, a0n, a1n, b0n, b1n);
    mma_frag(a0c, a1c, b0c, b1c, acc);
    a0c = a0n; a1c = a1n;
#pragma unroll
    for (int j = 0; j < 4; ++j) { b0c[j] = b0n[j]; b1c[j] = b1n[j]; }
  }
  mma_frag(a0c, a1c, b0c, b1c, acc);

  // C/D layout: VGPR r -> (M = mt*16 + 8*half + r, N = col)
  const int rowbase = (mt << 4) + (half << 3);
#pragma unroll
  for (int j = 0; j < 4; ++j) {
    const int col = (ng << 6) + (j << 4) + l16;
    const float bv = bias[col];
#pragma unroll
    for (int r = 0; r < 8; ++r) {
      float v = acc[j][r] + bv;
      if (relu) v = v > 0.f ? v : 0.f;
      size_t off = (size_t)(rowbase + r) * N + col;
      if (outH) Ch[off] = (_Float16)v; else Cf[off] = v;
    }
  }
}

// ============================================================
// max-pool over group dim: feat f16 (G*K x C) -> out f32 (G x C)
// ============================================================
__global__ void k_maxpool(const _Float16* __restrict__ feat, float* __restrict__ out,
                          int G, int K, int C) {
  int gid = blockIdx.x * blockDim.x + threadIdx.x;
  if (gid >= G * C) return;
  int g = gid / C, c = gid % C;
  const _Float16* f = feat + (size_t)g * K * C + c;
  float m = -3.0e38f;
  for (int k = 0; k < K; ++k) {
    float v = (float)f[(size_t)k * C];
    m = v > m ? v : m;
  }
  out[gid] = m;
}

// ============================================================
// 3-NN inverse-distance interpolation of 3-channel features
// ============================================================
__global__ void k_interp3(const float* __restrict__ qxyz, const float* __restrict__ rxyz,
                          const float* __restrict__ rfeat, float* __restrict__ out,
                          int M, int N) {
  int gid = blockIdx.x * blockDim.x + threadIdx.x;
  if (gid >= BB * M) return;
  int b = gid / M;
  const float* q = qxyz + (size_t)gid * 3;
  const float* p = rxyz + (size_t)b * N * 3;
  float qx = q[0], qy = q[1], qz = q[2];
  float d0 = 3e38f, d1 = 3e38f, d2 = 3e38f;
  int i0 = 0, i1 = 0, i2 = 0;
  for (int i = 0; i < N; ++i) {
    float dx = p[i * 3] - qx, dy = p[i * 3 + 1] - qy, dz = p[i * 3 + 2] - qz;
    float d = dx * dx + dy * dy + dz * dz;
    if (d < d0)      { d2 = d1; i2 = i1; d1 = d0; i1 = i0; d0 = d; i0 = i; }
    else if (d < d1) { d2 = d1; i2 = i1; d1 = d; i1 = i; }
    else if (d < d2) { d2 = d; i2 = i; }
  }
  float r0 = 1.f / (d0 + 1e-8f), r1 = 1.f / (d1 + 1e-8f), r2 = 1.f / (d2 + 1e-8f);
  float inv = 1.f / (r0 + r1 + r2);
  float w0 = r0 * inv, w1 = r1 * inv, w2 = r2 * inv;
  const float* f0 = rfeat + ((size_t)b * N + i0) * 3;
  const float* f1 = rfeat + ((size_t)b * N + i1) * 3;
  const float* f2 = rfeat + ((size_t)b * N + i2) * 3;
  float* o = out + (size_t)gid * 3;
  o[0] = w0 * f0[0] + w1 * f1[0] + w2 * f2[0];
  o[1] = w0 * f0[1] + w1 * f1[1] + w2 * f2[1];
  o[2] = w0 * f0[2] + w1 * f1[2] + w2 * f2[2];
}

// out[g, i*C + j] = sim[g, i] * feat[g, j]   (i = 0..2)
__global__ void k_outer(const float* __restrict__ sim, const float* __restrict__ feat,
                        float* __restrict__ out, int G, int C) {
  int gid = blockIdx.x * blockDim.x + threadIdx.x;
  if (gid >= G * C) return;
  int g = gid / C, j = gid % C;
  float f = feat[gid];
  const float* s = sim + (size_t)g * 3;
  float* o = out + (size_t)g * 3 * C + j;
  o[0]             = s[0] * f;
  o[(size_t)C]     = s[1] * f;
  o[2 * (size_t)C] = s[2] * f;
}

__global__ void k_f2h(const float* __restrict__ in, _Float16* __restrict__ out, int n) {
  int gid = blockIdx.x * blockDim.x + threadIdx.x;
  if (gid < n) out[gid] = (_Float16)in[gid];
}

// ============================================================
// host orchestration
// ============================================================
extern "C" void kernel_launch(void* const* d_in, const int* in_sizes, int n_in,
                              void* d_out, int out_size, void* d_ws, size_t ws_size,
                              hipStream_t stream) {
  (void)in_sizes; (void)n_in; (void)out_size; (void)ws_size;
  const float* xyz = (const float*)d_in[0];
  const float* Wp[11]; const float* Bp[11];
  for (int i = 0; i < 11; ++i) { Wp[i] = (const float*)d_in[1 + 2 * i]; Bp[i] = (const float*)d_in[2 + 2 * i]; }

  // ---- workspace bump allocator ----
  char* ws = (char*)d_ws;
  size_t off = 0;
  auto alloc = [&](size_t bytes) -> void* {
    void* p = ws + off;
    off = (off + bytes + 255) & ~(size_t)255;
    return p;
  };
  const int M1 = BB * S1 * NS1;       // 262144
  const int M2 = BB * S2 * NS2;       // 131072
  const int M3 = BB * S2;             // 2048
  const int KP1 = 32, KP2 = 416, KP3 = 800;

  float* coords   = (float*)alloc((size_t)BB * NN * 3 * 4);
  float* sim      = (float*)alloc((size_t)BB * NN * 3 * 4);
  float* dists    = (float*)alloc((size_t)BB * NN * 4);
  int*   fps1     = (int*)alloc((size_t)BB * S1 * 4);
  float* l1_xyz   = (float*)alloc((size_t)BB * S1 * 3 * 4);
  int*   bidx1    = (int*)alloc((size_t)BB * S1 * NS1 * 4);
  int*   fps2     = (int*)alloc((size_t)BB * S2 * 4);
  float* l2_xyz   = (float*)alloc((size_t)BB * S2 * 3 * 4);
  int*   bidx2    = (int*)alloc((size_t)BB * S2 * NS2 * 4);
  _Float16* Xbuf  = (_Float16*)alloc((size_t)M2 * KP2 * 2);      // largest X
  _Float16* actA  = (_Float16*)alloc((size_t)M2 * 384 * 2);      // largest act
  _Float16* actB  = (_Float16*)alloc((size_t)M2 * 384 * 2);
  _Float16* Wh    = (_Float16*)alloc((size_t)1024 * 1536 * 2);   // largest padded W
  float* l1_feat  = (float*)alloc((size_t)BB * S1 * 128 * 4);
  float* sim1     = (float*)alloc((size_t)BB * S1 * 3 * 4);
  float* l1_pts   = (float*)alloc((size_t)BB * S1 * 384 * 4);
  float* l2_feat  = (float*)alloc((size_t)BB * S2 * 256 * 4);
  float* sim2     = (float*)alloc((size_t)BB * S2 * 3 * 4);
  float* l2_pts   = (float*)alloc((size_t)BB * S2 * 768 * 4);
  float* l3_f     = (float*)alloc((size_t)BB * 1024 * 4);
  _Float16* l3_h  = (_Float16*)alloc((size_t)BB * 1024 * 2);
  _Float16* fc1_h = (_Float16*)alloc((size_t)BB * 512 * 2);

  auto cdiv = [](int a, int b) { return (a + b - 1) / b; };
  auto gemm = [&](const _Float16* A, int li, int M, int N, int Kin, int Kpad,
                  bool relu, _Float16* outH, float* outF) {
    int wtot = N * Kpad;
    k_wpad<<<cdiv(wtot, 256), 256, 0, stream>>>(Wp[li], Wh, N, Kin, Kpad);
    int tiles = (M / 16) * (N / 64);            // 16x64 strips per wave
    k_gemm_wmma<<<cdiv(tiles, 8), 256, 0, stream>>>(A, Wh, Bp[li], outH, outF,
                                                    M, N, Kpad, relu ? 1 : 0,
                                                    outH ? 1 : 0);
  };

  // ---- split input ----
  k_split<<<cdiv(BB * NN, 256), 256, 0, stream>>>(xyz, coords, sim);

  // ---- SA1 ----
  k_fps<<<BB, FPST, 0, stream>>>(coords, dists, fps1, NN, S1);
  k_gather3<<<cdiv(BB * S1, 256), 256, 0, stream>>>(coords, fps1, l1_xyz, NN, S1);
  k_ball<<<cdiv(BB * S1, 256), 256, 0, stream>>>(l1_xyz, coords, bidx1, S1, NN, NS1, 0.2f * 0.2f);
  k_group1<<<cdiv(M1, 256), 256, 0, stream>>>(coords, l1_xyz, bidx1, Xbuf, S1, NN, NS1, KP1);
  gemm(Xbuf, 0, M1, 64, 3, KP1, true, actA, nullptr);
  gemm(actA, 1, M1, 64, 64, 64, true, actB, nullptr);
  gemm(actB, 2, M1, 128, 64, 64, true, actA, nullptr);
  k_maxpool<<<cdiv(BB * S1 * 128, 256), 256, 0, stream>>>(actA, l1_feat, BB * S1, NS1, 128);
  k_interp3<<<cdiv(BB * S1, 256), 256, 0, stream>>>(l1_xyz, coords, sim, sim1, S1, NN);
  k_outer<<<cdiv(BB * S1 * 128, 256), 256, 0, stream>>>(sim1, l1_feat, l1_pts, BB * S1, 128);

  // ---- SA2 ----
  k_fps<<<BB, FPST, 0, stream>>>(l1_xyz, dists, fps2, S1, S2);
  k_gather3<<<cdiv(BB * S2, 256), 256, 0, stream>>>(l1_xyz, fps2, l2_xyz, S1, S2);
  k_ball<<<cdiv(BB * S2, 256), 256, 0, stream>>>(l2_xyz, l1_xyz, bidx2, S2, S1, NS2, 0.4f * 0.4f);
  k_group2<<<cdiv(M2, 256), 256, 0, stream>>>(l1_xyz, l1_pts, l2_xyz, bidx2, Xbuf, S1, S2, NS2, 384, KP2);
  gemm(Xbuf, 3, M2, 384, 387, KP2, true, actA, nullptr);
  gemm(actA, 4, M2, 384, 384, 384, true, actB, nullptr);
  gemm(actB, 5, M2, 256, 384, 384, true, actA, nullptr);
  k_maxpool<<<cdiv(BB * S2 * 256, 256), 256, 0, stream>>>(actA, l2_feat, BB * S2, NS2, 256);
  k_interp3<<<cdiv(BB * S2, 256), 256, 0, stream>>>(l2_xyz, coords, sim, sim2, S2, NN);
  k_outer<<<cdiv(BB * S2 * 256, 256), 256, 0, stream>>>(sim2, l2_feat, l2_pts, BB * S2, 256);

  // ---- SA3 (group-all) ----
  k_build3<<<cdiv(M3, 256), 256, 0, stream>>>(l2_xyz, l2_pts, Xbuf, M3, 768, KP3);
  gemm(Xbuf, 6, M3, 768, 771, KP3, true, actA, nullptr);
  gemm(actA, 7, M3, 1536, 768, 768, true, actB, nullptr);
  gemm(actB, 8, M3, 1024, 1536, 1536, true, actA, nullptr);
  k_maxpool<<<cdiv(BB * 1024, 256), 256, 0, stream>>>(actA, l3_f, BB, S2, 1024);
  k_f2h<<<cdiv(BB * 1024, 256), 256, 0, stream>>>(l3_f, l3_h, BB * 1024);

  // ---- FC head ----
  gemm(l3_h, 9, BB, 512, 1024, 1024, true, fc1_h, nullptr);
  gemm(fc1_h, 10, BB, 256, 512, 512, false, nullptr, (float*)d_out);
}
